// SpatialMaskAttention_78400333021798
// MI455X (gfx1250) — matrix operand order repaired and checked by reading, not compiled, not run on hardware
//
#include <hip/hip_runtime.h>
#include <hip/hip_bf16.h>

typedef __attribute__((ext_vector_type(16))) _Float16 v16h;
typedef __attribute__((ext_vector_type(8)))  _Float16 v8h;
typedef __attribute__((ext_vector_type(8)))  float    v8f;

#define BT  96
#define NN  512
#define FF  64
#define FIN 128
#define DD  64
#define NH  8
#define HD  8

// padded LDS strides (halves) to break bank conflicts on b128 accesses
#define XS_STRIDE 136   // proj activations (FIN=128 + 8)
#define SC_STRIDE 520   // score rows (NN=512 + 8)
#define FC_STRIDE 72    // fc tiles (DD=64 + 8)

__device__ __forceinline__ v8h zero8() {
  v8h z = {(_Float16)0, (_Float16)0, (_Float16)0, (_Float16)0,
           (_Float16)0, (_Float16)0, (_Float16)0, (_Float16)0};
  return z;
}
__device__ __forceinline__ v16h cat8(v8h lo, v8h hi) {
  return __builtin_shufflevector(lo, hi, 0, 1, 2, 3, 4, 5, 6, 7,
                                 8, 9, 10, 11, 12, 13, 14, 15);
}
__device__ __forceinline__ v8h ldv8(const _Float16* p) { return *(const v8h*)p; }

// ---------------- prep kernels ----------------

__global__ void prep_x_kernel(const float* __restrict__ X,
                              const float* __restrict__ STE,
                              _Float16* __restrict__ xh) {
  int idx = blockIdx.x * 256 + threadIdx.x;
  if (idx >= BT * NN * FIN) return;
  int c = idx & (FIN - 1);
  int row = idx >> 7;  // bt*N + n
  float v = (c < FF) ? X[row * FF + c] : STE[row * FF + (c - FF)];
  xh[idx] = (_Float16)v;
}

// weights stored TRANSPOSED ([col][K]) so B fragments are contiguous b128 loads
__global__ void prep_w_kernel(const float* __restrict__ WQ, const float* __restrict__ WK,
                              const float* __restrict__ WV, const float* __restrict__ bQ,
                              const float* __restrict__ bK, const float* __restrict__ bV,
                              const float* __restrict__ W1, const float* __restrict__ W2,
                              const int* __restrict__ adj,
                              _Float16* __restrict__ wqkvT, float* __restrict__ bqkv,
                              _Float16* __restrict__ w1T, _Float16* __restrict__ w2T,
                              unsigned* __restrict__ maskbits) {
  int idx = blockIdx.x * 256 + threadIdx.x;
  if (idx < 192 * 128) {           // wqkvT[c][r] = W*(r, c)
    int c = idx >> 7, r = idx & 127;
    float v = (c < 64) ? WQ[r * 64 + c] : (c < 128) ? WK[r * 64 + c - 64] : WV[r * 64 + c - 128];
    wqkvT[idx] = (_Float16)v;
    return;
  }
  idx -= 192 * 128;
  if (idx < 192) {
    bqkv[idx] = (idx < 64) ? bQ[idx] : (idx < 128) ? bK[idx - 64] : bV[idx - 128];
    return;
  }
  idx -= 192;
  if (idx < 4096) { int c = idx >> 6, r = idx & 63; w1T[idx] = (_Float16)W1[r * 64 + c]; return; }
  idx -= 4096;
  if (idx < 4096) { int c = idx >> 6, r = idx & 63; w2T[idx] = (_Float16)W2[r * 64 + c]; return; }
  idx -= 4096;
  if (idx < NN * 16) {
    int i = idx >> 4, seg = idx & 15;
    unsigned w = 0;
    for (int b = 0; b < 32; ++b)
      if (adj[i * NN + seg * 32 + b] > 0) w |= (1u << b);
    maskbits[idx] = w;
  }
}

// ---------------- QKV projection: x[64x128] @ Wqkv[128x192] + b, relu ----------------

__global__ void proj_kernel(const _Float16* __restrict__ xh,
                            const _Float16* __restrict__ wqkvT,
                            const float* __restrict__ bqkv,
                            _Float16* __restrict__ qkv) {
  extern __shared__ char smem[];
  _Float16* xs  = (_Float16*)smem;             // 64 rows x XS_STRIDE
  _Float16* wsm = xs + 64 * XS_STRIDE;         // 192 rows(cols) x XS_STRIDE
  int bt = blockIdx.y, nt = blockIdx.x;
  int tid = threadIdx.x;
  const _Float16* xsrc = xh + (size_t)(bt * NN + nt * 64) * FIN;
  for (int c = tid; c < 64 * 16; c += 256) {     // 16B chunks, padded rows
    int r = c >> 4, w = c & 15;
    ((uint4*)(xs + r * XS_STRIDE))[w] = ((const uint4*)(xsrc + r * FIN))[w];
  }
  for (int c = tid; c < 192 * 16; c += 256) {
    int r = c >> 4, w = c & 15;
    ((uint4*)(wsm + r * XS_STRIDE))[w] = ((const uint4*)(wqkvT + r * FIN))[w];
  }
  __syncthreads();
  int wave = tid >> 5, lane = tid & 31, g = lane >> 4, ln = lane & 15;
  for (int t = wave; t < 48; t += 8) {   // 4 row-tiles x 12 col-tiles
    int tr = t / 12, tc = t % 12;
    v8f acc = {0.f, 0.f, 0.f, 0.f, 0.f, 0.f, 0.f, 0.f};
    for (int kb = 0; kb < 4; ++kb) {
      const _Float16* ap = xs + (tr * 16 + ln) * XS_STRIDE + kb * 32 + g * 16;
      const _Float16* bp = wsm + (tc * 16 + ln) * XS_STRIDE + kb * 32 + g * 16;
      v16h a = cat8(ldv8(ap), ldv8(ap + 8));
      v16h b = cat8(ldv8(bp), ldv8(bp + 8));
      acc = __builtin_amdgcn_wmma_f32_16x16x32_f16(false, a, false, b, (short)0, acc, false, false);
    }
    int col = tc * 16 + ln;
    float bias = bqkv[col];
#pragma unroll
    for (int j = 0; j < 8; ++j) {
      int m = g * 8 + j;
      int row = nt * 64 + tr * 16 + m;
      float v = acc[j] + bias;
      v = v > 0.f ? v : 0.f;
      qkv[(size_t)(bt * NN + row) * 192 + col] = (_Float16)v;
    }
  }
}

// ---------------- attention: per (bt, head, 64-query block), 4 waves ----------------

__global__ void attn_kernel(const _Float16* __restrict__ qkv,
                            const unsigned* __restrict__ maskbits,
                            _Float16* __restrict__ aoh) {
  extern __shared__ char smem[];
  _Float16* kls = (_Float16*)smem;                 // [512][8]  (== B layout for QK^T)
  _Float16* vt  = kls + NN * HD;                   // [8][SC_STRIDE] (V transposed)
  _Float16* sc  = vt + HD * SC_STRIDE;             // 4 waves x 16 rows x SC_STRIDE
  unsigned* mwords = (unsigned*)(sc + 4 * 16 * SC_STRIDE);  // [64][16]
  float* linv = (float*)(mwords + 64 * 16);        // 4*16
  int qb = blockIdx.x, head = blockIdx.y, bt = blockIdx.z;
  int tid = threadIdx.x;
  for (int t = tid; t < NN * HD; t += 128) {
    int j = t >> 3, dd = t & 7;
    size_t base = (size_t)(bt * NN + j) * 192 + head * HD + dd;
    kls[t] = qkv[base + 64];
    vt[dd * SC_STRIDE + j] = qkv[base + 128];
  }
  for (int t = tid; t < 64 * 16; t += 128)
    mwords[t] = maskbits[(qb * 64 + (t >> 4)) * 16 + (t & 15)];
  int wave = tid >> 5, lane = tid & 31, g = lane >> 4, ln = lane & 15;
  int i0 = qb * 64 + wave * 16;
  // Q fragment: rows i0+ln, K = d (real only for group 0's low chunk)
  v8h qlo = zero8();
  if (g == 0) qlo = *(const v8h*)(qkv + (size_t)(bt * NN + i0 + ln) * 192 + head * HD);
  v16h aq = cat8(qlo, zero8());
  __syncthreads();
  _Float16* scw = sc + wave * 16 * SC_STRIDE;
  const float scale = 0.35355339059327373f;  // 1/sqrt(8)
  unsigned mw[8];
  // pass 1: scores + mask, staged to LDS as f16
  for (int jb = 0; jb < 32; ++jb) {
    v8h klo = zero8();
    if (g == 0) klo = ldv8(kls + (jb * 16 + ln) * HD);
    v16h b = cat8(klo, zero8());
    v8f acc = {0.f, 0.f, 0.f, 0.f, 0.f, 0.f, 0.f, 0.f};
    acc = __builtin_amdgcn_wmma_f32_16x16x32_f16(false, aq, false, b, (short)0, acc, false, false);
    if ((jb & 1) == 0) {           // seg = (jb*16+ln)>>5 == jb>>1 exactly
      int seg = jb >> 1;
#pragma unroll
      for (int j = 0; j < 8; ++j) mw[j] = mwords[(wave * 16 + g * 8 + j) * 16 + seg];
    }
    int jg = jb * 16 + ln;
    int bit = (jb & 1) * 16 + ln;
#pragma unroll
    for (int j = 0; j < 8; ++j) {
      int m = g * 8 + j;
      float s = ((mw[j] >> bit) & 1u) ? acc[j] * scale : -32767.0f;
      scw[m * SC_STRIDE + jg] = (_Float16)s;
    }
  }
  __syncthreads();
  // pass 2: row softmax; each lane owns a contiguous 16-element chunk of the row
  for (int r = 0; r < 16; ++r) {
    _Float16* rowp = scw + r * SC_STRIDE + lane * 16;
    v8h p0 = ldv8(rowp), p1 = ldv8(rowp + 8);
    float vv[16];
#pragma unroll
    for (int c = 0; c < 8; ++c) { vv[c] = (float)p0[c]; vv[c + 8] = (float)p1[c]; }
    float mx = vv[0];
#pragma unroll
    for (int c = 1; c < 16; ++c) mx = fmaxf(mx, vv[c]);
#pragma unroll
    for (int off = 16; off > 0; off >>= 1) mx = fmaxf(mx, __shfl_xor(mx, off, 32));
    float sum = 0.f;
    v8h e0, e1;
#pragma unroll
    for (int c = 0; c < 8; ++c) {
      float pa = __expf(vv[c] - mx);
      float pb = __expf(vv[c + 8] - mx);
      e0[c] = (_Float16)pa; e1[c] = (_Float16)pb;
      sum += pa + pb;
    }
    *(v8h*)rowp = e0;
    *(v8h*)(rowp + 8) = e1;
#pragma unroll
    for (int off = 16; off > 0; off >>= 1) sum += __shfl_xor(sum, off, 32);
    if (lane == 0) linv[wave * 16 + r] = 1.0f / sum;
  }
  __syncthreads();
  // pass 3: O = P @ V (d padded to 16 cols), scale rows by 1/l
  v8f o = {0.f, 0.f, 0.f, 0.f, 0.f, 0.f, 0.f, 0.f};
  for (int kb = 0; kb < 16; ++kb) {
    const _Float16* ap = scw + ln * SC_STRIDE + kb * 32 + g * 16;
    v16h a = cat8(ldv8(ap), ldv8(ap + 8));
    v8h b0 = zero8(), b1 = zero8();
    if (ln < HD) {
      const _Float16* vp = vt + ln * SC_STRIDE + kb * 32 + g * 16;
      b0 = ldv8(vp); b1 = ldv8(vp + 8);
    }
    v16h b = cat8(b0, b1);
    o = __builtin_amdgcn_wmma_f32_16x16x32_f16(false, a, false, b, (short)0, o, false, false);
  }
  if (ln < HD) {
#pragma unroll
    for (int j = 0; j < 8; ++j) {
      int m = g * 8 + j;
      float val = o[j] * linv[wave * 16 + m];
      aoh[(size_t)(bt * NN + i0 + m) * DD + head * HD + ln] = (_Float16)val;
    }
  }
}

// ---------------- FC head: relu(ao @ W1 + b1) @ W2 + b2 ----------------

__global__ void fc_kernel(const _Float16* __restrict__ aoh,
                          const _Float16* __restrict__ w1T,
                          const _Float16* __restrict__ w2T,
                          const float* __restrict__ b1,
                          const float* __restrict__ b2,
                          float* __restrict__ out) {
  extern __shared__ char smem[];
  _Float16* aos = (_Float16*)smem;                 // 128 x FC_STRIDE
  _Float16* w1s = aos + 128 * FC_STRIDE;           // 64 x FC_STRIDE (transposed)
  _Float16* w2s = w1s + 64 * FC_STRIDE;            // 64 x FC_STRIDE (transposed)
  _Float16* hid = w2s + 64 * FC_STRIDE;            // 128 x FC_STRIDE
  int bt = blockIdx.y, rt = blockIdx.x;
  int tid = threadIdx.x;
  const _Float16* src = aoh + (size_t)(bt * NN + rt * 128) * DD;
  for (int c = tid; c < 128 * 8; c += 256) {
    int r = c >> 3, w = c & 7;
    ((uint4*)(aos + r * FC_STRIDE))[w] = ((const uint4*)(src + r * DD))[w];
  }
  for (int c = tid; c < 64 * 8; c += 256) {
    int r = c >> 3, w = c & 7;
    ((uint4*)(w1s + r * FC_STRIDE))[w] = ((const uint4*)(w1T + r * DD))[w];
    ((uint4*)(w2s + r * FC_STRIDE))[w] = ((const uint4*)(w2T + r * DD))[w];
  }
  __syncthreads();
  int wave = tid >> 5, lane = tid & 31, g = lane >> 4, ln = lane & 15;
  for (int t = wave; t < 32; t += 8) {   // 8 row-tiles x 4 col-tiles
    int tr = t >> 2, tc = t & 3;
    v8f acc = {0.f, 0.f, 0.f, 0.f, 0.f, 0.f, 0.f, 0.f};
    for (int kb = 0; kb < 2; ++kb) {
      const _Float16* ap = aos + (tr * 16 + ln) * FC_STRIDE + kb * 32 + g * 16;
      const _Float16* bp = w1s + (tc * 16 + ln) * FC_STRIDE + kb * 32 + g * 16;
      v16h a = cat8(ldv8(ap), ldv8(ap + 8));
      v16h b = cat8(ldv8(bp), ldv8(bp + 8));
      acc = __builtin_amdgcn_wmma_f32_16x16x32_f16(false, a, false, b, (short)0, acc, false, false);
    }
    float bias = b1[tc * 16 + ln];
#pragma unroll
    for (int j = 0; j < 8; ++j) {
      int m = g * 8 + j;
      float v = acc[j] + bias;
      v = v > 0.f ? v : 0.f;
      hid[(tr * 16 + m) * FC_STRIDE + tc * 16 + ln] = (_Float16)v;
    }
  }
  __syncthreads();
  for (int t = wave; t < 32; t += 8) {
    int tr = t >> 2, tc = t & 3;
    v8f acc = {0.f, 0.f, 0.f, 0.f, 0.f, 0.f, 0.f, 0.f};
    for (int kb = 0; kb < 2; ++kb) {
      const _Float16* ap = hid + (tr * 16 + ln) * FC_STRIDE + kb * 32 + g * 16;
      const _Float16* bp = w2s + (tc * 16 + ln) * FC_STRIDE + kb * 32 + g * 16;
      v16h a = cat8(ldv8(ap), ldv8(ap + 8));
      v16h b = cat8(ldv8(bp), ldv8(bp + 8));
      acc = __builtin_amdgcn_wmma_f32_16x16x32_f16(false, a, false, b, (short)0, acc, false, false);
    }
    float bias = b2[tc * 16 + ln];
#pragma unroll
    for (int j = 0; j < 8; ++j) {
      int m = g * 8 + j;
      out[(size_t)(bt * NN + rt * 128 + tr * 16 + m) * DD + tc * 16 + ln] = acc[j] + bias;
    }
  }
}

// ---------------- launcher ----------------

extern "C" void kernel_launch(void* const* d_in, const int* in_sizes, int n_in,
                              void* d_out, int out_size, void* d_ws, size_t ws_size,
                              hipStream_t stream) {
  const float* X   = (const float*)d_in[0];
  const float* STE = (const float*)d_in[1];
  const int*   adj = (const int*)d_in[2];
  const float* WQ  = (const float*)d_in[3];
  const float* bQ  = (const float*)d_in[4];
  const float* WK  = (const float*)d_in[5];
  const float* bK  = (const float*)d_in[6];
  const float* WV  = (const float*)d_in[7];
  const float* bV  = (const float*)d_in[8];
  const float* W1  = (const float*)d_in[9];
  const float* b1  = (const float*)d_in[10];
  const float* W2  = (const float*)d_in[11];
  const float* b2  = (const float*)d_in[12];
  float* out = (float*)d_out;

  char* ws = (char*)d_ws;
  size_t off = 0;
  auto alloc = [&](size_t bytes) {
    char* p = ws + off;
    off = (off + bytes + 255) & ~(size_t)255;
    return p;
  };
  _Float16* xh    = (_Float16*)alloc((size_t)BT * NN * FIN * 2);
  _Float16* wqkvT = (_Float16*)alloc(192 * 128 * 2);
  float*    bqkv  = (float*)alloc(192 * 4);
  _Float16* w1T   = (_Float16*)alloc(64 * 64 * 2);
  _Float16* w2T   = (_Float16*)alloc(64 * 64 * 2);
  unsigned* mb    = (unsigned*)alloc(NN * 16 * 4);
  _Float16* qkv   = (_Float16*)alloc((size_t)BT * NN * 192 * 2);
  _Float16* aoh   = (_Float16*)alloc((size_t)BT * NN * DD * 2);
  (void)ws_size; (void)in_sizes; (void)n_in; (void)out_size;

  prep_x_kernel<<<(BT * NN * FIN + 255) / 256, 256, 0, stream>>>(X, STE, xh);
  int prep_items = 192 * 128 + 192 + 4096 + 4096 + NN * 16;
  prep_w_kernel<<<(prep_items + 255) / 256, 256, 0, stream>>>(WQ, WK, WV, bQ, bK, bV,
                                                              W1, W2, adj, wqkvT, bqkv,
                                                              w1T, w2T, mb);
  size_t proj_smem = (size_t)(64 + 192) * XS_STRIDE * 2;  // ~68 KB
  proj_kernel<<<dim3(8, BT), 256, proj_smem, stream>>>(xh, wqkvT, bqkv, qkv);
  size_t attn_smem = (size_t)NN * HD * 2 + (size_t)HD * SC_STRIDE * 2 +
                     (size_t)4 * 16 * SC_STRIDE * 2 + 64 * 16 * 4 + 4 * 16 * 4;  // ~87 KB
  attn_kernel<<<dim3(8, NH, BT), 128, attn_smem, stream>>>(qkv, mb, aoh);
  size_t fc_smem = (size_t)(128 + 64 + 64 + 128) * FC_STRIDE * 2;  // ~55 KB
  fc_kernel<<<dim3(4, BT), 256, fc_smem, stream>>>(aoh, w1T, w2T, b1, b2, out);
}
